// PolicyNetwork_32255204393673
// MI455X (gfx1250) — matrix operand order repaired and checked by reading, not compiled, not run on hardware
//
#include <hip/hip_runtime.h>
#include <hip/hip_bf16.h>

typedef __attribute__((ext_vector_type(16))) _Float16 v16h;
typedef __attribute__((ext_vector_type(8)))  _Float16 v8h;
typedef __attribute__((ext_vector_type(8)))  float    v8f;
typedef __attribute__((ext_vector_type(4)))  float    v4f;

#define HID 128

// ---------------------------------------------------------------------------
// Utility kernels
// ---------------------------------------------------------------------------
__global__ void fill_zero_kernel(float* __restrict__ p, int n) {
  int i = blockIdx.x * 256 + threadIdx.x;
  if (i < n) p[i] = 0.0f;
}

__global__ void relu_kernel(float* __restrict__ p, int n) {
  int i = blockIdx.x * 256 + threadIdx.x;
  if (i < n) p[i] = fmaxf(p[i], 0.0f);
}

// ---------------------------------------------------------------------------
// Weight pre-pack: fp32 [K,128] row-major -> f16 WMMA B fragments.
// Fragment f = ct*(K/32)+kc ; element ((f*32+lane)*16 + j) holds
//   B[k = kc*32 + (j>=8)*16 + (lane>=16)*8 + (j&7)][n = ct*16 + (lane&15)]
// matching the CDNA5 16-bit B-matrix VGPR layout (K in 8-value groups).
// ---------------------------------------------------------------------------
__global__ void pack_w_kernel(const float* __restrict__ W,
                              _Float16* __restrict__ Wf, int K) {
  int gid = blockIdx.x * 256 + threadIdx.x;
  int kch = K >> 5;
  int total = 8 * kch * 32 * 16;
  if (gid >= total) return;
  int j    = gid & 15;
  int lane = (gid >> 4) & 31;
  int f    = gid >> 9;
  int kc   = f % kch;
  int ct   = f / kch;
  int k    = kc * 32 + (j >> 3) * 16 + (lane >> 4) * 8 + (j & 7);
  int n    = ct * 16 + (lane & 15);
  Wf[gid] = (_Float16)W[(size_t)k * HID + n];
}

// ---------------------------------------------------------------------------
// WMMA GEMM: C[N,128] = act( A[N,K] @ W[K,128] + bias ), K in {64,128}.
// Block = 256 threads = 8 waves; block stages its 16-row A tile in LDS
// *already converted to f16* (stride K+8 halves: 16B-aligned rows, lanes
// land 4 banks apart -> conflict-free ds_load_b128).  Each wave owns a
// 16-col tile and runs v_wmma_f32_16x16x32_f16 with f32 accumulate.
// In-place safe: blocks are row-disjoint and A is consumed from LDS.
// ---------------------------------------------------------------------------
__global__ __launch_bounds__(256)
void wmma_gemm_kernel(const float* __restrict__ A,
                      const _Float16* __restrict__ Wf,
                      const float* __restrict__ bias,
                      float* __restrict__ C, int K, int relu_out) {
  __shared__ _Float16 Ash[16 * 136];  // 16 rows, stride K+8 halves (<=136)
  const int t    = threadIdx.x;
  const int lane = t & 31;
  const int wave = t >> 5;
  const int row0 = blockIdx.x << 4;
  const int kch  = K >> 5;
  const int ldk  = K + 8;

  // stage + convert: fp32 global -> f16 LDS (done once per block)
  for (int idx = t; idx < 16 * K; idx += 256) {
    int r = idx / K;
    int c = idx - r * K;
    Ash[r * ldk + c] = (_Float16)A[(size_t)(row0 + r) * K + c];
  }
  __syncthreads();

  // A fragment: lane<16 -> M=lane, K = kc*32 + {0..7,16..23}
  //             lane>=16 -> M=lane-16, K = kc*32 + {8..15,24..31}
  const int m  = lane & 15;
  const int hi = lane >> 4;
  const _Float16* ar = Ash + m * ldk + hi * 8;

  v8f acc = {};
  for (int kc = 0; kc < kch; ++kc) {
    const _Float16* ap = ar + kc * 32;
    v8h l0 = *(const v8h*)(ap);        // 16B ds_load_b128
    v8h l1 = *(const v8h*)(ap + 16);   // 16B ds_load_b128
    v16h af;
#pragma unroll
    for (int j = 0; j < 8; ++j) {
      af[j]     = l0[j];
      af[8 + j] = l1[j];
    }
    v16h bf = *(const v16h*)(Wf + ((size_t)(wave * kch + kc) * 32 + lane) * 16);
    acc = __builtin_amdgcn_wmma_f32_16x16x32_f16(
        /*neg_a=*/false, af, /*neg_b=*/false, bf,
        /*c_mod=*/(short)0, acc, /*reuse_a=*/false, /*reuse_b=*/false);
  }

  // C layout: lanes 0-15 -> M=r, N=lane ; lanes 16-31 -> M=8+r, N=lane-16
  const int n   = lane & 15;
  const int col = (wave << 4) + n;
  const float b = bias[col];
  const int mb  = row0 + hi * 8;
#pragma unroll
  for (int r = 0; r < 8; ++r) {
    float v = acc[r] + b;
    if (relu_out) v = fmaxf(v, 0.0f);
    C[(size_t)(mb + r) * HID + col] = v;
  }
}

// ---------------------------------------------------------------------------
// Degree count over rows (edges + self loops), then deg -> deg^-0.5 in place
// ---------------------------------------------------------------------------
__global__ void degree_kernel(const int* __restrict__ row,
                              float* __restrict__ deg, int E_, int total) {
  int idx = blockIdx.x * 256 + threadIdx.x;
  if (idx >= total) return;
  int r = (idx < E_) ? row[idx] : (idx - E_);
  unsafeAtomicAdd(deg + r, 1.0f);
}

__global__ void invsqrt_kernel(float* __restrict__ deg, int n) {
  int i = blockIdx.x * 256 + threadIdx.x;
  if (i < n) {
    float d = deg[i];
    deg[i] = (d > 0.0f) ? rsqrtf(d) : 0.0f;
  }
}

// ---------------------------------------------------------------------------
// GCN aggregation: out[col] += dis[row]*dis[col] * xw[row], one wave/message,
// 32 lanes x float4 = 128 cols, hardware f32 atomics into the target row.
// ---------------------------------------------------------------------------
__global__ __launch_bounds__(256)
void aggregate_kernel(const float* __restrict__ xw, const float* __restrict__ dis,
                      const int* __restrict__ row, const int* __restrict__ col,
                      float* __restrict__ out, int E_, int total) {
  int lane = threadIdx.x & 31;
  int idx  = blockIdx.x * 8 + (threadIdx.x >> 5);
  if (idx >= total) return;
  int r, c;
  if (idx < E_) { r = row[idx]; c = col[idx]; }
  else          { r = c = idx - E_; }
  float nrm = dis[r] * dis[c];
  v4f v = ((const v4f*)(xw + (size_t)r * HID))[lane];
  float* o = out + (size_t)c * HID + lane * 4;
  unsafeAtomicAdd(o + 0, nrm * v[0]);
  unsafeAtomicAdd(o + 1, nrm * v[1]);
  unsafeAtomicAdd(o + 2, nrm * v[2]);
  unsafeAtomicAdd(o + 3, nrm * v[3]);
}

// ---------------------------------------------------------------------------
// Column sum of h[N,128] into gsum[128] (block-local accumulate, few atomics)
// ---------------------------------------------------------------------------
__global__ __launch_bounds__(256)
void colsum_kernel(const float* __restrict__ h, float* __restrict__ gsum,
                   int nrows) {
  __shared__ float sd[256];
  int t    = threadIdx.x;
  int c    = t & 127;
  int rsel = t >> 7;
  int r0   = blockIdx.x * 256;
  float acc = 0.0f;
  int rend = min(r0 + 256, nrows);
  for (int r = r0 + rsel; r < rend; r += 2) acc += h[(size_t)r * HID + c];
  sd[t] = acc;
  __syncthreads();
  if (t < 128) unsafeAtomicAdd(gsum + t, sd[t] + sd[t + 128]);
}

// ---------------------------------------------------------------------------
// Global head: g = relu(mean(h)@W_g + b_g); emits end_score, state_value and
// the scalar s_g = g @ W_a[128:256] + b_a reused by the per-node scores.
// ---------------------------------------------------------------------------
__global__ __launch_bounds__(128)
void global_head_kernel(const float* __restrict__ gsum,
                        const float* __restrict__ Wg, const float* __restrict__ bg,
                        const float* __restrict__ Wa, const float* __restrict__ ba,
                        const float* __restrict__ Wv, const float* __restrict__ bv,
                        float* __restrict__ sg_out, float* __restrict__ out,
                        int nnodes) {
  __shared__ float sg[128];
  __shared__ float red[3 * 128];
  int t = threadIdx.x;
  sg[t] = gsum[t] * (1.0f / (float)nnodes);
  __syncthreads();
  float acc = bg[t];
  for (int k = 0; k < 128; ++k) acc += sg[k] * Wg[k * HID + t];
  float gv = fmaxf(acc, 0.0f);
  red[t]       = gv * Wa[t];        // node-part weight
  red[128 + t] = gv * Wa[128 + t];  // global-part weight
  red[256 + t] = gv * Wv[t];
  __syncthreads();
  if (t == 0) {
    float s1 = 0.f, s2 = 0.f, s3 = 0.f;
    for (int k = 0; k < 128; ++k) {
      s1 += red[k]; s2 += red[128 + k]; s3 += red[256 + k];
    }
    sg_out[0]       = s2 + ba[0];       // g @ Wa[128:] + b_a
    out[nnodes]     = s1 + s2 + ba[0];  // end score: concat(g,g)@Wa + b_a
    out[nnodes + 1] = s3 + bv[0];       // state value
  }
}

// ---------------------------------------------------------------------------
// Per-node action scores: one wave per node, float4 lanes, wave32 shfl reduce
// ---------------------------------------------------------------------------
__global__ __launch_bounds__(256)
void node_scores_kernel(const float* __restrict__ h, const float* __restrict__ Wa,
                        const float* __restrict__ sg, float* __restrict__ out,
                        int nnodes) {
  int lane = threadIdx.x & 31;
  int idx  = blockIdx.x * 8 + (threadIdx.x >> 5);
  if (idx >= nnodes) return;
  v4f hv = ((const v4f*)(h + (size_t)idx * HID))[lane];
  v4f wv = ((const v4f*)Wa)[lane];
  float p = hv[0] * wv[0] + hv[1] * wv[1] + hv[2] * wv[2] + hv[3] * wv[3];
#pragma unroll
  for (int off = 16; off > 0; off >>= 1) p += __shfl_xor(p, off, 32);
  if (lane == 0) out[idx] = p + sg[0];
}

// ---------------------------------------------------------------------------
// Launcher
// ---------------------------------------------------------------------------
extern "C" void kernel_launch(void* const* d_in, const int* in_sizes, int n_in,
                              void* d_out, int out_size, void* d_ws, size_t ws_size,
                              hipStream_t stream) {
  const float* x    = (const float*)d_in[0];
  const int*   ei   = (const int*)d_in[1];
  const float* W_e1 = (const float*)d_in[2];
  const float* b_e1 = (const float*)d_in[3];
  const float* W_e2 = (const float*)d_in[4];
  const float* b_e2 = (const float*)d_in[5];
  const float* W_c1 = (const float*)d_in[6];
  const float* b_c1 = (const float*)d_in[7];
  const float* W_c2 = (const float*)d_in[8];
  const float* b_c2 = (const float*)d_in[9];
  const float* W_g  = (const float*)d_in[10];
  const float* b_g  = (const float*)d_in[11];
  const float* W_a  = (const float*)d_in[12];
  const float* b_a  = (const float*)d_in[13];
  const float* W_v  = (const float*)d_in[14];
  const float* b_v  = (const float*)d_in[15];

  const int IN  = 64;
  const int N   = in_sizes[0] / IN;     // 50000
  const int E   = in_sizes[1] / 2;      // 800000
  const int TOT = E + N;                // edges + self loops
  const int* row = ei;
  const int* col = ei + E;
  float* out = (float*)d_out;

  // workspace layout (floats)
  float* buf0 = (float*)d_ws;             // N*128 ping
  float* buf1 = buf0 + (size_t)N * HID;   // N*128 pong
  float* dis  = buf1 + (size_t)N * HID;   // N (degree, then deg^-1/2)
  float* gsum = dis + N;                  // 128
  float* sgp  = gsum + 128;               // 16 (scalar + pad)
  _Float16* wf_e1 = (_Float16*)(sgp + 16);   // 8*2*32*16  halves (K=64)
  _Float16* wf_e2 = wf_e1 + 8192;            // 8*4*32*16  halves (K=128)
  _Float16* wf_c1 = wf_e2 + 16384;
  _Float16* wf_c2 = wf_c1 + 16384;

  const int nElems = N * HID;
  const int gE = (nElems + 255) / 256;
  const int gRowTiles = N / 16;  // 50000 % 16 == 0
  const int gMsg = (TOT + 7) / 8;

  // 1) pack weights into f16 WMMA B fragments
  pack_w_kernel<<<32, 256, 0, stream>>>(W_e1, wf_e1, 64);
  pack_w_kernel<<<64, 256, 0, stream>>>(W_e2, wf_e2, 128);
  pack_w_kernel<<<64, 256, 0, stream>>>(W_c1, wf_c1, 128);
  pack_w_kernel<<<64, 256, 0, stream>>>(W_c2, wf_c2, 128);

  // 2) symmetric-normalization coefficients
  fill_zero_kernel<<<(N + 255) / 256, 256, 0, stream>>>(dis, N);
  degree_kernel<<<(TOT + 255) / 256, 256, 0, stream>>>(row, dis, E, TOT);
  invsqrt_kernel<<<(N + 255) / 256, 256, 0, stream>>>(dis, N);

  // 3) node embed: buf0 = relu(x@We1+b) ; buf0 = buf0@We2+b (in place)
  wmma_gemm_kernel<<<gRowTiles, 256, 0, stream>>>(x,    wf_e1, b_e1, buf0, 64,  1);
  wmma_gemm_kernel<<<gRowTiles, 256, 0, stream>>>(buf0, wf_e2, b_e2, buf0, 128, 0);

  // 4) conv1: xw in place in buf0, aggregate into buf1, relu
  wmma_gemm_kernel<<<gRowTiles, 256, 0, stream>>>(buf0, wf_c1, b_c1, buf0, 128, 0);
  fill_zero_kernel<<<gE, 256, 0, stream>>>(buf1, nElems);
  aggregate_kernel<<<gMsg, 256, 0, stream>>>(buf0, dis, row, col, buf1, E, TOT);
  relu_kernel<<<gE, 256, 0, stream>>>(buf1, nElems);

  // 5) conv2: xw in place in buf1, aggregate into buf0, relu -> final h
  wmma_gemm_kernel<<<gRowTiles, 256, 0, stream>>>(buf1, wf_c2, b_c2, buf1, 128, 0);
  fill_zero_kernel<<<gE, 256, 0, stream>>>(buf0, nElems);
  aggregate_kernel<<<gMsg, 256, 0, stream>>>(buf1, dis, row, col, buf0, E, TOT);
  relu_kernel<<<gE, 256, 0, stream>>>(buf0, nElems);

  // 6) global embed + heads
  fill_zero_kernel<<<1, 256, 0, stream>>>(gsum, 128);
  colsum_kernel<<<(N + 255) / 256, 256, 0, stream>>>(buf0, gsum, N);
  global_head_kernel<<<1, 128, 0, stream>>>(gsum, W_g, b_g, W_a, b_a, W_v, b_v,
                                            sgp, out, N);
  node_scores_kernel<<<(N + 7) / 8, 256, 0, stream>>>(buf0, W_a, sgp, out, N);
}
